// GritTransformerLayer_74466142978756
// MI455X (gfx1250) — compile-verified, matrix operand order checked by reading, not run on hardware
//
#include <hip/hip_runtime.h>
#include <hip/hip_bf16.h>

// ---------------------------------------------------------------------------
// CDNA5 (gfx1250) bf16 WMMA transformer layer.
// All matmuls use v_wmma_f32_16x16x32_bf16 (wave32, 16x16 C tiles, fp32 acc).
// Fragment layout per CDNA5 ISA (16-bit A 16x32 / B 32x16):
//   lane L in [0,15]:  row/col = L,    regs v0-3 = K k0..k0+7,  v4-7 = k0+16..k0+23
//   lane L in [16,31]: row/col = L-16, regs v0-3 = K k0+8..+15, v4-7 = k0+24..+31
// => per-lane fragment = two contiguous 16B loads (b128) from a K-contiguous row.
//
// The dominant memory stream is attn_bias (512 MB, read once -> ~22us HBM
// floor at 23.3 TB/s). The attention kernel double-buffers its bias tiles
// through the CDNA5 async global->LDS path (ASYNCcnt-tracked), plus
// global_prefetch hints two tiles ahead.
// ---------------------------------------------------------------------------

#if __has_builtin(__builtin_amdgcn_global_load_async_to_lds_b128)
#define HAVE_ASYNC_LDS 1
#else
#define HAVE_ASYNC_LDS 0
#endif

typedef __attribute__((ext_vector_type(16))) __bf16 v16bf;
typedef __attribute__((ext_vector_type(8)))  float  v8f;

__device__ inline unsigned short f2bf(float f) {
  union { float f; unsigned int u; } x; x.f = f;
  unsigned int r = x.u + 0x7FFFu + ((x.u >> 16) & 1u);   // round-to-nearest-even
  return (unsigned short)(r >> 16);
}

union FragU { v16bf v; uint4 q[2]; };

// rowbase = &buf[row * ld]; K contiguous. k0 multiple of 32, koff = (lane>>4)*8.
__device__ inline v16bf load_frag(const unsigned short* rowbase, int k0, int koff) {
  FragU f;
  f.q[0] = *(const uint4*)(rowbase + k0 + koff);
  f.q[1] = *(const uint4*)(rowbase + k0 + 16 + koff);
  return f.v;
}

__device__ inline v8f wmma_bf16(v16bf a, v16bf b, v8f c) {
  return __builtin_amdgcn_wmma_f32_16x16x32_bf16(false, a, false, b, (short)0, c,
                                                 false, false);
}

#if HAVE_ASYNC_LDS
// 16B async copy global -> LDS (ASYNCcnt-tracked). The builtin expects
// "int __vector(4) AS1*" / "AS3*" params (per hipcc diagnostic), so rebuild
// the pointers with explicit address spaces via integer round-trip: AS1 keeps
// the 64-bit value; AS3 is the 32-bit LDS offset = low 32 bits of the generic
// LDS address (CDNA5 ISA: LDS aperture truncates addr[31:0]).
typedef int v4i_b __attribute__((vector_size(16)));
typedef __attribute__((address_space(1))) v4i_b gas_v4i;
typedef __attribute__((address_space(3))) v4i_b las_v4i;

__device__ inline void async_ld16(const void* gsrc, void* ldst) {
  __builtin_amdgcn_global_load_async_to_lds_b128(
      (gas_v4i*)(uintptr_t)gsrc,
      (las_v4i*)(unsigned int)(uintptr_t)ldst, 0, 0);
}
template <int CNT>
__device__ inline void wait_async() {
#if __has_builtin(__builtin_amdgcn_s_wait_asynccnt)
  __builtin_amdgcn_s_wait_asynccnt(CNT);
#else
  if (CNT == 0) asm volatile("s_wait_asynccnt 0x0" ::: "memory");
  else          asm volatile("s_wait_asynccnt 0x4" ::: "memory");
#endif
  asm volatile("" ::: "memory");
}
#endif

// ---------------------------------------------------------------------------
// Format conversion kernels
// ---------------------------------------------------------------------------
__global__ __launch_bounds__(256) void k_f32_to_bf16(const float* __restrict__ in,
                                                     unsigned short* __restrict__ out,
                                                     int n) {
  int i = blockIdx.x * blockDim.x + threadIdx.x;
  if (i < n) out[i] = f2bf(in[i]);
}

// W [K, Nout] fp32 row-major  ->  Wt [Nout, K] bf16 (K contiguous)
__global__ __launch_bounds__(256) void k_transpose_bf16(const float* __restrict__ W,
                                                        unsigned short* __restrict__ Wt,
                                                        int K, int Nout) {
  int i = blockIdx.x * blockDim.x + threadIdx.x;
  if (i >= K * Nout) return;
  int n = i / K, k = i - n * K;
  Wt[i] = f2bf(W[(size_t)k * Nout + n]);
}

// ---------------------------------------------------------------------------
// Generic WMMA GEMM:  C[M,Nout] = A[M,K](bf16) * Wt[Nout,K](bf16)^T + bias
// One wave = 16x64 output tile (4 WMMA tiles sharing the A fragment).
// mode: 0 = bf16 [M,Nout]; 1 = bf16 [H,M,64] (h=n>>6); 2 = bf16 [H,64,M] (V^T)
// ---------------------------------------------------------------------------
__global__ __launch_bounds__(256) void k_gemm_bf16(
    const unsigned short* __restrict__ A, const unsigned short* __restrict__ Wt,
    const float* __restrict__ bias, unsigned short* __restrict__ out,
    int M, int Nout, int K, int mode, int relu) {
  int lane = threadIdx.x & 31;
  int gw = blockIdx.x * (blockDim.x >> 5) + (threadIdx.x >> 5);
  int nTiles = Nout >> 6;
  int mTile = gw / nTiles, nTile = gw - mTile * nTiles;
  if (mTile * 16 >= M) return;
  int row = lane & 15, hi = lane >> 4, koff = hi << 3;

  const unsigned short* Arow = A + (size_t)(mTile * 16 + row) * K;
  const unsigned short* Brow = Wt + (size_t)(nTile * 64 + row) * K;

  v8f acc[4];
#pragma unroll
  for (int t = 0; t < 4; ++t)
#pragma unroll
    for (int r = 0; r < 8; ++r) acc[t][r] = 0.0f;

  for (int k0 = 0; k0 < K; k0 += 32) {
    v16bf a = load_frag(Arow, k0, koff);
#pragma unroll
    for (int t = 0; t < 4; ++t) {
      v16bf b = load_frag(Brow + (size_t)t * 16 * K, k0, koff);
      acc[t] = wmma_bf16(a, b, acc[t]);
    }
  }

  int nl = lane & 15;
#pragma unroll
  for (int t = 0; t < 4; ++t) {
    int nbase = nTile * 64 + t * 16;
    float bv = bias[nbase + nl];
#pragma unroll
    for (int r = 0; r < 8; ++r) {
      float val = acc[t][r] + bv;
      if (relu) val = fmaxf(val, 0.0f);
      int m = mTile * 16 + r + hi * 8;
      int n = nbase + nl;
      size_t idx;
      if (mode == 0) idx = (size_t)m * Nout + n;
      else if (mode == 1) { int h = n >> 6, tt = n & 63; idx = ((size_t)h * M + m) * 64 + tt; }
      else               { int h = n >> 6, tt = n & 63; idx = ((size_t)h * 64 + tt) * M + m; }
      out[idx] = f2bf(val);
    }
  }
}

// C[M,Nout](fp32) = A*Wt^T + bias + res   (residual epilogue for O-proj / FFN2)
__global__ __launch_bounds__(256) void k_gemm_f32res(
    const unsigned short* __restrict__ A, const unsigned short* __restrict__ Wt,
    const float* __restrict__ bias, const float* __restrict__ res,
    float* __restrict__ out, int M, int Nout, int K) {
  int lane = threadIdx.x & 31;
  int gw = blockIdx.x * (blockDim.x >> 5) + (threadIdx.x >> 5);
  int nTiles = Nout >> 6;
  int mTile = gw / nTiles, nTile = gw - mTile * nTiles;
  if (mTile * 16 >= M) return;
  int row = lane & 15, hi = lane >> 4, koff = hi << 3;

  const unsigned short* Arow = A + (size_t)(mTile * 16 + row) * K;
  const unsigned short* Brow = Wt + (size_t)(nTile * 64 + row) * K;

  v8f acc[4];
#pragma unroll
  for (int t = 0; t < 4; ++t)
#pragma unroll
    for (int r = 0; r < 8; ++r) acc[t][r] = 0.0f;

  for (int k0 = 0; k0 < K; k0 += 32) {
    v16bf a = load_frag(Arow, k0, koff);
#pragma unroll
    for (int t = 0; t < 4; ++t) {
      v16bf b = load_frag(Brow + (size_t)t * 16 * K, k0, koff);
      acc[t] = wmma_bf16(a, b, acc[t]);
    }
  }

  int nl = lane & 15;
#pragma unroll
  for (int t = 0; t < 4; ++t) {
    int nbase = nTile * 64 + t * 16;
    float bv = bias[nbase + nl];
#pragma unroll
    for (int r = 0; r < 8; ++r) {
      int m = mTile * 16 + r + hi * 8;
      size_t idx = (size_t)m * Nout + nbase + nl;
      out[idx] = acc[t][r] + bv + res[idx];
    }
  }
}

// ---------------------------------------------------------------------------
// Flash attention: one wave owns 16 query rows of one head, sweeps keys in
// 32-wide blocks with online softmax. The 512MB attn_bias stream is staged
// through LDS with double buffering (async global->LDS when available) so
// tile j+32 is in flight while tile j feeds the WMMAs.
// Q,K: [H,N,64] bf16.  Vt: [H,64,N] bf16.  out: [N,512] bf16.
// ---------------------------------------------------------------------------
__global__ __launch_bounds__(256) void k_attention(
    const unsigned short* __restrict__ Q, const unsigned short* __restrict__ Km,
    const unsigned short* __restrict__ Vt, const float* __restrict__ bias,
    unsigned short* __restrict__ out, int N) {
  const int HD = 64, H = 8;
  __shared__ __align__(16) unsigned short Pld[8][16 * 32];
#if HAVE_ASYNC_LDS
  __shared__ __align__(16) float Bld[8][2][16 * 32];
#else
  __shared__ __align__(16) float Bld[8][1][16 * 32];
#endif

  int lane = threadIdx.x & 31;
  int wv = threadIdx.x >> 5;
  int gw = blockIdx.x * 8 + wv;
  int qTiles = N >> 4;
  int h = gw / qTiles;
  int qt = gw - h * qTiles;
  if (h >= H) return;
  int row = lane & 15, hi = lane >> 4, koff = hi << 3;

  const unsigned short* Qh = Q + ((size_t)h * N + qt * 16 + row) * HD;
  const unsigned short* Kh = Km + (size_t)h * N * HD;
  const unsigned short* Vh = Vt + (size_t)h * HD * N;
  const float* Bh = bias + (size_t)h * N * N + (size_t)(qt * 16) * N;

  v16bf qa0 = load_frag(Qh, 0, koff);
  v16bf qa1 = load_frag(Qh, 32, koff);

  v8f o0, o1, o2, o3;
  float mrun[8], lrun[8];
#pragma unroll
  for (int r = 0; r < 8; ++r) {
    o0[r] = 0.f; o1[r] = 0.f; o2[r] = 0.f; o3[r] = 0.f;
    mrun[r] = -3.0e38f; lrun[r] = 0.f;
  }

  unsigned short* Pw = Pld[wv];

#if HAVE_ASYNC_LDS
  // prologue: async-fetch bias tile 0 into buffer 0 (4 x lane-b128 = 2KB tile)
  {
    int rr = lane >> 3, cc = (lane & 7) * 4;
#pragma unroll
    for (int i = 0; i < 4; ++i)
      async_ld16(Bh + (size_t)(4 * i + rr) * N + cc,
                 (char*)&Bld[wv][0][0] + (size_t)i * 512 + lane * 16);
  }
#else
  float breg[16];
#pragma unroll
  for (int i = 0; i < 16; ++i) breg[i] = Bh[(size_t)i * N + lane];
#endif

  for (int j = 0; j < N; j += 32) {
#if HAVE_ASYNC_LDS
    int cur = (j >> 5) & 1;
    float* Bw = &Bld[wv][cur][0];
    if (j + 32 < N) {   // async-fetch next tile into the other buffer
      int rr = lane >> 3, cc = (lane & 7) * 4;
#pragma unroll
      for (int i = 0; i < 4; ++i)
        async_ld16(Bh + (size_t)(4 * i + rr) * N + (j + 32) + cc,
                   (char*)&Bld[wv][cur ^ 1][0] + (size_t)i * 512 + lane * 16);
      wait_async<4>();  // in-order completion: current tile's 4 are done
    } else {
      wait_async<0>();
    }
#else
    float* Bw = &Bld[wv][0][0];
    // stage current tile (loaded last iteration into registers) to LDS
#pragma unroll
    for (int i = 0; i < 16; ++i) Bw[i * 32 + lane] = breg[i];
    asm volatile("s_wait_dscnt 0x0" ::: "memory");
    if (j + 32 < N) {   // start next tile's global loads (consumed next iter)
#pragma unroll
      for (int i = 0; i < 16; ++i) breg[i] = Bh[(size_t)i * N + (j + 32) + lane];
    }
#endif
    // L2-warming prefetch two tiles ahead (global_prefetch_b8; 16 lanes = 16 lines)
    if (j + 64 < N) __builtin_prefetch(Bh + (size_t)(lane & 15) * N + (j + 64), 0, 1);

    // scores S = Q K^T for keys [j, j+32)
    v8f s0, s1;
#pragma unroll
    for (int r = 0; r < 8; ++r) { s0[r] = 0.f; s1[r] = 0.f; }
    const unsigned short* K0 = Kh + (size_t)(j + row) * HD;
    s0 = wmma_bf16(qa0, load_frag(K0, 0, koff), s0);
    s0 = wmma_bf16(qa1, load_frag(K0, 32, koff), s0);
    const unsigned short* K1 = K0 + 16 * HD;
    s1 = wmma_bf16(qa0, load_frag(K1, 0, koff), s1);
    s1 = wmma_bf16(qa1, load_frag(K1, 32, koff), s1);

    // online softmax (row = r + 8*hi lives on a 16-lane group)
#pragma unroll
    for (int r = 0; r < 8; ++r) {
      int m = r + hi * 8;
      float a0 = s0[r] * 0.125f + Bw[m * 32 + row];
      float a1 = s1[r] * 0.125f + Bw[m * 32 + 16 + row];
      float t = fmaxf(a0, a1);
      t = fmaxf(t, __shfl_xor(t, 1, 32));
      t = fmaxf(t, __shfl_xor(t, 2, 32));
      t = fmaxf(t, __shfl_xor(t, 4, 32));
      t = fmaxf(t, __shfl_xor(t, 8, 32));
      float mn = fmaxf(mrun[r], t);
      float corr = __expf(mrun[r] - mn);
      float p0 = __expf(a0 - mn);
      float p1 = __expf(a1 - mn);
      float ps = p0 + p1;
      ps += __shfl_xor(ps, 1, 32);
      ps += __shfl_xor(ps, 2, 32);
      ps += __shfl_xor(ps, 4, 32);
      ps += __shfl_xor(ps, 8, 32);
      lrun[r] = lrun[r] * corr + ps;
      mrun[r] = mn;
      o0[r] *= corr; o1[r] *= corr; o2[r] *= corr; o3[r] *= corr;
      Pw[m * 32 + row]      = f2bf(p0);    // P tile 16x32 row-major in LDS
      Pw[m * 32 + 16 + row] = f2bf(p1);
    }
    asm volatile("s_wait_dscnt 0x0" ::: "memory");

    // O += P(16x32) @ V(32x64): A-fragment reload of P from LDS, B from Vt rows
    v16bf pa = load_frag(Pw + row * 32, 0, koff);
    o0 = wmma_bf16(pa, load_frag(Vh + (size_t)(0 * 16 + row) * N, j, koff), o0);
    o1 = wmma_bf16(pa, load_frag(Vh + (size_t)(1 * 16 + row) * N, j, koff), o1);
    o2 = wmma_bf16(pa, load_frag(Vh + (size_t)(2 * 16 + row) * N, j, koff), o2);
    o3 = wmma_bf16(pa, load_frag(Vh + (size_t)(3 * 16 + row) * N, j, koff), o3);
  }

  // epilogue: normalize and scatter into [N, 512] bf16 (head-interleaved)
#pragma unroll
  for (int r = 0; r < 8; ++r) {
    int m = qt * 16 + r + hi * 8;
    float inv = 1.0f / lrun[r];
    size_t base = (size_t)m * 512 + h * 64 + (lane & 15);
    out[base + 0]  = f2bf(o0[r] * inv);
    out[base + 16] = f2bf(o1[r] * inv);
    out[base + 32] = f2bf(o2[r] * inv);
    out[base + 48] = f2bf(o3[r] * inv);
  }
}

// ---------------------------------------------------------------------------
// LayerNorm over d=512, one block (8 waves) per row. Writes fp32 (+optional bf16).
// ---------------------------------------------------------------------------
__global__ __launch_bounds__(256) void k_layernorm(
    const float* __restrict__ y, const float* __restrict__ g, const float* __restrict__ b,
    float* __restrict__ xout, unsigned short* __restrict__ xbf, int d) {
  int rowi = blockIdx.x;
  const float* yr = y + (size_t)rowi * d;
  int tid = threadIdx.x;
  float v0 = yr[tid];
  float v1 = yr[tid + 256];
  float s = v0 + v1;
  float s2 = v0 * v0 + v1 * v1;
#pragma unroll
  for (int off = 1; off < 32; off <<= 1) {
    s += __shfl_xor(s, off, 32);
    s2 += __shfl_xor(s2, off, 32);
  }
  __shared__ float sh[16];
  int wv = tid >> 5;
  if ((tid & 31) == 0) { sh[wv] = s; sh[wv + 8] = s2; }
  __syncthreads();
  float ts = 0.f, ts2 = 0.f;
#pragma unroll
  for (int i = 0; i < 8; ++i) { ts += sh[i]; ts2 += sh[i + 8]; }
  float mean = ts / (float)d;
  float var = ts2 / (float)d - mean * mean;
  float rstd = rsqrtf(var + 1e-5f);
  float o0 = (v0 - mean) * rstd * g[tid] + b[tid];
  float o1 = (v1 - mean) * rstd * g[tid + 256] + b[tid + 256];
  size_t base = (size_t)rowi * d;
  xout[base + tid] = o0;
  xout[base + tid + 256] = o1;
  if (xbf) {
    xbf[base + tid] = f2bf(o0);
    xbf[base + tid + 256] = f2bf(o1);
  }
}

// ---------------------------------------------------------------------------
// Host orchestration
// ---------------------------------------------------------------------------
extern "C" void kernel_launch(void* const* d_in, const int* in_sizes, int n_in,
                              void* d_out, int out_size, void* d_ws, size_t ws_size,
                              hipStream_t stream) {
  (void)in_sizes; (void)n_in; (void)out_size; (void)ws_size;
  const int N = 4096, D = 512, H = 8, DF = 1024;

  const float* x  = (const float*)d_in[0];
  const float* ab = (const float*)d_in[1];
  const float* Wq = (const float*)d_in[2];
  const float* bq = (const float*)d_in[3];
  const float* Wk = (const float*)d_in[4];
  const float* bk = (const float*)d_in[5];
  const float* Wv = (const float*)d_in[6];
  const float* bv = (const float*)d_in[7];
  const float* Wo = (const float*)d_in[8];
  const float* bo = (const float*)d_in[9];
  const float* g1 = (const float*)d_in[10];
  const float* b1 = (const float*)d_in[11];
  const float* g2 = (const float*)d_in[12];
  const float* b2 = (const float*)d_in[13];
  const float* W1 = (const float*)d_in[14];
  const float* c1 = (const float*)d_in[15];
  const float* W2 = (const float*)d_in[16];
  const float* c2 = (const float*)d_in[17];
  float* outp = (float*)d_out;

  char* w = (char*)d_ws;
  size_t off = 0;
  auto alloc = [&](size_t bytes) -> void* {
    void* p = w + off;
    off = (off + bytes + 255) & ~(size_t)255;
    return p;
  };
  unsigned short* xbf  = (unsigned short*)alloc((size_t)N * D * 2);
  unsigned short* Wqt  = (unsigned short*)alloc((size_t)D * D * 2);
  unsigned short* Wkt  = (unsigned short*)alloc((size_t)D * D * 2);
  unsigned short* Wvt  = (unsigned short*)alloc((size_t)D * D * 2);
  unsigned short* Wot  = (unsigned short*)alloc((size_t)D * D * 2);
  unsigned short* W1t  = (unsigned short*)alloc((size_t)D * DF * 2);
  unsigned short* W2t  = (unsigned short*)alloc((size_t)DF * D * 2);
  unsigned short* Qb   = (unsigned short*)alloc((size_t)N * D * 2);   // [H,N,64]
  unsigned short* Kb   = (unsigned short*)alloc((size_t)N * D * 2);   // [H,N,64]
  unsigned short* Vtb  = (unsigned short*)alloc((size_t)N * D * 2);   // [H,64,N]
  unsigned short* attb = (unsigned short*)alloc((size_t)N * D * 2);   // [N,512]
  float*          ybuf = (float*)alloc((size_t)N * D * 4);
  float*          x1   = (float*)alloc((size_t)N * D * 4);
  unsigned short* x1bf = (unsigned short*)alloc((size_t)N * D * 2);
  unsigned short* hbuf = (unsigned short*)alloc((size_t)N * DF * 2);
  float*          y2   = (float*)alloc((size_t)N * D * 4);

  const int T = 256;
  // precision prep: activations + weights to bf16 (weights transposed to [Nout,K])
  k_f32_to_bf16<<<(N * D) / T, T, 0, stream>>>(x, xbf, N * D);
  k_transpose_bf16<<<(D * D + T - 1) / T, T, 0, stream>>>(Wq, Wqt, D, D);
  k_transpose_bf16<<<(D * D + T - 1) / T, T, 0, stream>>>(Wk, Wkt, D, D);
  k_transpose_bf16<<<(D * D + T - 1) / T, T, 0, stream>>>(Wv, Wvt, D, D);
  k_transpose_bf16<<<(D * D + T - 1) / T, T, 0, stream>>>(Wo, Wot, D, D);
  k_transpose_bf16<<<(D * DF + T - 1) / T, T, 0, stream>>>(W1, W1t, D, DF);
  k_transpose_bf16<<<(DF * D + T - 1) / T, T, 0, stream>>>(W2, W2t, DF, D);

  // QKV projections (wave tile 16x64): waves = (N/16)*(D/64) = 2048 -> 256 blocks
  int qkvBlocks = ((N / 16) * (D / 64)) / 8;
  k_gemm_bf16<<<qkvBlocks, T, 0, stream>>>(xbf, Wqt, bq, Qb, N, D, D, 1, 0);
  k_gemm_bf16<<<qkvBlocks, T, 0, stream>>>(xbf, Wkt, bk, Kb, N, D, D, 1, 0);
  k_gemm_bf16<<<qkvBlocks, T, 0, stream>>>(xbf, Wvt, bv, Vtb, N, D, D, 2, 0);

  // flash attention: waves = H*(N/16) = 2048 -> 256 blocks
  k_attention<<<(H * (N / 16)) / 8, T, 0, stream>>>(Qb, Kb, Vtb, ab, attb, N);

  // O projection + residual, LN1
  k_gemm_f32res<<<qkvBlocks, T, 0, stream>>>(attb, Wot, bo, x, ybuf, N, D, D);
  k_layernorm<<<N, T, 0, stream>>>(ybuf, g1, b1, x1, x1bf, D);

  // FFN1 (relu, bf16 out): waves = (N/16)*(DF/64) = 4096 -> 512 blocks
  int ffn1Blocks = ((N / 16) * (DF / 64)) / 8;
  k_gemm_bf16<<<ffn1Blocks, T, 0, stream>>>(x1bf, W1t, c1, hbuf, N, DF, D, 0, 1);

  // FFN2 + residual, LN2 -> final output
  k_gemm_f32res<<<qkvBlocks, T, 0, stream>>>(hbuf, W2t, c2, x1, y2, N, D, DF);
  k_layernorm<<<N, T, 0, stream>>>(y2, g2, b2, outp, nullptr, D);
}
